// CVMultiheadAttention_35966056137411
// MI455X (gfx1250) — matrix operand order repaired and checked by reading, not compiled, not run on hardware
//
#include <hip/hip_runtime.h>
#include <hip/hip_bf16.h>

// ---------------------------------------------------------------------------
// CDNA5 (gfx1250) complex-valued multi-head attention, bf16 WMMA pipeline
// with async global->LDS staging (ASYNCcnt), double buffering, swizzled LDS
// layouts, and explicit 128-bit fragment loads (no scalar repacking).
// ---------------------------------------------------------------------------

typedef __attribute__((ext_vector_type(8)))  float   v8f;
typedef __attribute__((ext_vector_type(16))) __bf16  v16bf;

typedef unsigned short bf16_t;

#define CB  4
#define CS  1024
#define CD  1024
#define CH  16
#define CDK 64
#define CDV 64
#define CEPS 1e-6f

union FragBF {
    v16bf v;
    uint4 q[2];
    unsigned short u[16];
};

// Load a fragment as two explicit 16B vector loads (base must be 32B aligned).
// Forces ds_load_b128 / global_load_b128 pairs with no 16-bit repacking.
__device__ __forceinline__ void load_frag(FragBF& f, const bf16_t* p) {
    const uint4* q = (const uint4*)p;
    f.q[0] = q[0];
    f.q[1] = q[1];
}

__device__ __forceinline__ bf16_t f2bf(float f) {
    unsigned u = __float_as_uint(f);
    unsigned r = (u + 0x7FFFu + ((u >> 16) & 1u)) >> 16;
    return (bf16_t)r;
}

// Swap bits 3<->4 of the element index within each 32-element K-block:
// stored order [0..7, 16..23, 8..15, 24..31]. A-fragment for half h then
// reads 16 CONTIGUOUS elements at offset h*16.
__device__ __forceinline__ int swz32(int c) {
    return (c & ~31) | (c & 7) | ((c & 8) << 1) | ((c & 16) >> 1);
}

__device__ __forceinline__ v8f wmma_bf16(const FragBF& a, const FragBF& b, v8f c) {
    // D = A(16x32 bf16) x B(32x16 bf16) + C(16x16 f32)
    return __builtin_amdgcn_wmma_f32_16x16x32_bf16(
        /*neg_a=*/false, a.v, /*neg_b=*/false, b.v,
        /*c_mod=*/(short)0, c, /*reuse_a=*/false, /*reuse_b=*/false);
}

// Per-lane async copy of 16 bytes global -> LDS (tracked by ASYNCcnt).
// VDST = LDS byte address (low 32 bits of flat shared addr), VADDR = 64b global.
__device__ __forceinline__ void async_g2l_b128(void* lds_p, const void* gptr) {
    unsigned loff = (unsigned)(size_t)lds_p;
    unsigned long long ga = (unsigned long long)(size_t)gptr;
    asm volatile("global_load_async_to_lds_b128 %0, %1, off"
                 :: "v"(loff), "v"(ga)
                 : "memory");
}

__device__ __forceinline__ void wait_async_le4() {
    asm volatile("s_wait_asynccnt 4" ::: "memory");
}
__device__ __forceinline__ void wait_async_0() {
    asm volatile("s_wait_asynccnt 0" ::: "memory");
}

// ---------------------------------------------------------------------------
// f32 -> bf16 conversion
// ---------------------------------------------------------------------------
__global__ __launch_bounds__(256) void k_f32_to_bf16(const float* __restrict__ in,
                                                     bf16_t* __restrict__ out, int n) {
    int i = blockIdx.x * 256 + threadIdx.x;
    if (i < n) out[i] = f2bf(in[i]);
}

// ---------------------------------------------------------------------------
// Complex GEMM:  C[M,N] = A[M,K] @ W[N,K]^T   (torch Linear convention)
//   Cr = Ar Wr^T - Ai Wi^T ; Ci = Ar Wi^T + Ai Wr^T
// mode: 0 = plain [M,N] store, 1 = head permute [B,H,S,Dh], 2 = transposed
//       head permute [B,H,Dh,S] (for V, so attention reads contiguous keys).
// 256 threads = 8 waves; block tile 64x64; each wave does 16x32.
// Staging: per-thread 16B async global->LDS b128 copies, double buffered.
// A tile stored K-swizzled so A-fragments are single contiguous 32B runs.
// ---------------------------------------------------------------------------
__global__ __launch_bounds__(256)
void k_cgemm_bf16(const bf16_t* __restrict__ Ar, const bf16_t* __restrict__ Ai,
                  const bf16_t* __restrict__ Wr, const bf16_t* __restrict__ Wi,
                  bf16_t* __restrict__ Obr, bf16_t* __restrict__ Obi,
                  float* __restrict__ Ofr, float* __restrict__ Ofi,
                  int M, int N, int K, int mode) {
    __shared__ bf16_t lA[2][2][64 * 32];   // [buf][r/i][64 rows x 32 k] (swizzled)
    __shared__ bf16_t lW[2][2][64 * 32];   // [buf][r/i][64 n-rows x 32 k]

    const int tid  = threadIdx.x;
    const int wave = tid >> 5;
    const int lane = tid & 31;
    const int half = lane >> 4;
    const int l15  = lane & 15;

    const int m0 = blockIdx.x * 64;
    const int n0 = blockIdx.y * 64;
    const int wm = (wave & 3) * 16;   // wave row offset in block tile
    const int wn = (wave >> 2) * 32;  // wave col offset in block tile

    // per-thread 8-element (16B) staging chunk
    const int idx  = tid * 8;          // 0..2047
    const int srow = idx >> 5, scol = idx & 31;
    const int pcol = swz32(scol);      // swizzled K position (A and W tiles)

    v8f accRR[2] = {v8f{}, v8f{}};
    v8f accII[2] = {v8f{}, v8f{}};
    v8f accRI[2] = {v8f{}, v8f{}};

    auto issue = [&](int buf, int k0) {
        size_t ga = (size_t)(m0 + srow) * K + k0 + scol;
        size_t gw = (size_t)(n0 + srow) * K + k0 + scol;
        async_g2l_b128(&lA[buf][0][srow * 32 + pcol], Ar + ga);
        async_g2l_b128(&lA[buf][1][srow * 32 + pcol], Ai + ga);
        async_g2l_b128(&lW[buf][0][srow * 32 + pcol], Wr + gw);
        async_g2l_b128(&lW[buf][1][srow * 32 + pcol], Wi + gw);
    };

    issue(0, 0);
    const int nk = K >> 5;
    for (int t = 0; t < nk; ++t) {
        const int cur = t & 1;
        __syncthreads();                  // everyone done reading buf cur^1
        const bool more = (t + 1) < nk;
        if (more) {
            issue(cur ^ 1, (t + 1) * 32);
            wait_async_le4();             // in-order: current buffer's 4 ops done
        } else {
            wait_async_0();
        }
        __syncthreads();                  // current buffer visible to all waves

        // A fragments: single contiguous 32B run (swizzled storage) -> 2x b128
        FragBF ar, ai;
        {
            int rbase = (wm + l15) * 32 + half * 16;
            load_frag(ar, &lA[cur][0][rbase]);
            load_frag(ai, &lA[cur][1][rbase]);
        }
#pragma unroll
        for (int nt = 0; nt < 2; ++nt) {
            // B fragments (32x16 of W^T): single contiguous 32B run -> 2x b128
            FragBF br, bi;
            int nbase = (wn + nt * 16 + l15) * 32 + half * 16;
            load_frag(br, &lW[cur][0][nbase]);
            load_frag(bi, &lW[cur][1][nbase]);
            accRR[nt] = wmma_bf16(ar, br, accRR[nt]);
            accII[nt] = wmma_bf16(ai, bi, accII[nt]);
            accRI[nt] = wmma_bf16(ar, bi, accRI[nt]);
            accRI[nt] = wmma_bf16(ai, br, accRI[nt]);
        }
    }

    // Epilogue: C/D layout -> VGPR v holds row (v + 8*half), col = l15.
#pragma unroll
    for (int nt = 0; nt < 2; ++nt) {
#pragma unroll
        for (int v = 0; v < 8; ++v) {
            int m = m0 + wm + v + 8 * half;
            int n = n0 + wn + nt * 16 + l15;
            float cr = accRR[nt][v] - accII[nt][v];
            float ci = accRI[nt][v];
            size_t oidx;
            if (mode == 1) {
                int b = m >> 10, s = m & 1023;      // S = 1024
                int h = n >> 6, d = n & 63;         // Dh = 64
                oidx = (((size_t)b * CH + h) * CS + s) * 64 + d;
            } else if (mode == 2) {
                int b = m >> 10, s = m & 1023;
                int h = n >> 6, d = n & 63;
                oidx = (((size_t)b * CH + h) * 64 + d) * CS + s;  // [B,H,Dh,S]
            } else {
                oidx = (size_t)m * N + n;
            }
            if (Obr) { Obr[oidx] = f2bf(cr); Obi[oidx] = f2bf(ci); }
            if (Ofr) { Ofr[oidx] = cr;       Ofi[oidx] = ci; }
        }
    }
}

// ---------------------------------------------------------------------------
// Fused attention: one workgroup per (b*H+h, 16-query block).
// Pass 1: WMMA scores -> per-row min/max of |attn| (LDS uint atomics).
// Pass 2: recompute scores, mag-min-max scale, LDS relayout to A-fragments,
//         WMMA attn@V, cross-wave reduce via ds_add_f32, bf16 store.
// Q/K bf16 [B,H,S,64]; V bf16 TRANSPOSED [B,H,64,S]; O bf16 [B,S,H*64].
// Q tile and score scratch stored in natural K order (contiguous fragment reads).
// ---------------------------------------------------------------------------
__global__ __launch_bounds__(256)
void k_cattention(const bf16_t* __restrict__ Qr, const bf16_t* __restrict__ Qi,
                  const bf16_t* __restrict__ Kr, const bf16_t* __restrict__ Ki,
                  const bf16_t* __restrict__ Vtr, const bf16_t* __restrict__ Vti,
                  bf16_t* __restrict__ Or_, bf16_t* __restrict__ Oi_) {
    __shared__ bf16_t   lQr[16 * 64];
    __shared__ bf16_t   lQi[16 * 64];
    __shared__ unsigned rowmin[16];
    __shared__ unsigned rowmax[16];
    __shared__ float    lOr[16 * 64];
    __shared__ float    lOi[16 * 64];
    __shared__ bf16_t   scr[8][2][16 * 32];   // per-wave scaled-score scratch

    const int bid  = blockIdx.x;
    const int qb   = bid & 63;     // query block (S/16 = 64)
    const int bh   = bid >> 6;     // b*H + h
    const size_t base  = (size_t)bh * CS * 64;  // Q/K per-head base
    const size_t baseT = (size_t)bh * 64 * CS;  // V^T per-head base

    const int tid  = threadIdx.x;
    const int wave = tid >> 5;
    const int lane = tid & 31;
    const int half = lane >> 4;
    const int l15  = lane & 15;

    if (tid < 16) { rowmin[tid] = 0x7F800000u; rowmax[tid] = 0u; }
    for (int i = tid; i < 16 * 64; i += 256) { lOr[i] = 0.0f; lOi[i] = 0.0f; }
    for (int i = tid; i < 16 * 64; i += 256) {
        int r = i >> 6, c = i & 63;
        size_t g = base + (size_t)(qb * 16 + r) * 64 + c;
        int cp = c;  // unswizzled (contiguous K order, matches K fragments)
        lQr[r * 64 + cp] = Qr[g];
        lQi[r * 64 + cp] = Qi[g];
    }
    __syncthreads();

    // Q A-fragments for the two K-chunks: contiguous 32B runs (unswizzled)
    FragBF qfr[2], qfi[2];
#pragma unroll
    for (int ch = 0; ch < 2; ++ch) {
        int qbase = l15 * 64 + ch * 32 + half * 16;
        load_frag(qfr[ch], &lQr[qbase]);
        load_frag(qfi[ch], &lQi[qbase]);
    }
    const float invt = 0.125f;  // 1/sqrt(DK)

    auto scoreTile = [&](int kt, v8f& srr, v8f& sii, v8f& sri) {
#pragma unroll
        for (int ch = 0; ch < 2; ++ch) {
            FragBF br, bi;  // K^T fragment: contiguous 32B of key row l15
            size_t krow = base + (size_t)(kt * 16 + l15) * 64 + ch * 32 + half * 16;
            load_frag(br, &Kr[krow]);
            load_frag(bi, &Ki[krow]);
            srr = wmma_bf16(qfr[ch], br, srr);
            sii = wmma_bf16(qfi[ch], bi, sii);
            sri = wmma_bf16(qfr[ch], bi, sri);
            sri = wmma_bf16(qfi[ch], br, sri);
        }
    };

    // ---- Pass 1: row-wise min/max of |attn| --------------------------------
    float rmn[8], rmx[8];
#pragma unroll
    for (int v = 0; v < 8; ++v) { rmn[v] = 3.402823466e38f; rmx[v] = 0.0f; }

    for (int kt = wave; kt < 64; kt += 8) {
        v8f srr = {}, sii = {}, sri = {};
        scoreTile(kt, srr, sii, sri);
#pragma unroll
        for (int v = 0; v < 8; ++v) {
            float sr = (srr[v] - sii[v]) * invt;
            float si = sri[v] * invt;
            float mag = sqrtf(sr * sr + si * si);
            rmn[v] = fminf(rmn[v], mag);
            rmx[v] = fmaxf(rmx[v], mag);
        }
    }
#pragma unroll
    for (int v = 0; v < 8; ++v) {
        int r = v + 8 * half;
        atomicMin(&rowmin[r], __float_as_uint(rmn[v]));  // mags >= 0: uint order OK
        atomicMax(&rowmax[r], __float_as_uint(rmx[v]));
    }
    __syncthreads();

    // ---- Pass 2: recompute, scale, attn @ V --------------------------------
    v8f oRR[4] = {v8f{}, v8f{}, v8f{}, v8f{}};
    v8f oII[4] = {v8f{}, v8f{}, v8f{}, v8f{}};
    v8f oRI[4] = {v8f{}, v8f{}, v8f{}, v8f{}};

    for (int kt2 = wave * 8; kt2 < wave * 8 + 8; kt2 += 2) {
#pragma unroll
        for (int s = 0; s < 2; ++s) {
            int kt = kt2 + s;
            v8f srr = {}, sii = {}, sri = {};
            scoreTile(kt, srr, sii, sri);
            // unswizzled column position for key-col (s*16 + l15)
            int cp = (s << 4) | l15;
#pragma unroll
            for (int v = 0; v < 8; ++v) {
                float sr = (srr[v] - sii[v]) * invt;
                float si = sri[v] * invt;
                float mag = sqrtf(sr * sr + si * si);
                int r = v + 8 * half;
                float mn = __uint_as_float(rowmin[r]);
                float mx = __uint_as_float(rowmax[r]);
                float sc = (mag - mn) / ((mx - mn) * mag + 1e-12f);
                scr[wave][0][r * 32 + cp] = f2bf(sr * sc);
                scr[wave][1][r * 32 + cp] = f2bf(si * sc);
            }
        }
        // wave-local LDS RAW: make the cross-lane relayout visible
        asm volatile("s_wait_dscnt 0" ::: "memory");

        FragBF pr, pi;  // scaled scores as A-fragment: contiguous 32B run
        {
            int pbase = l15 * 32 + half * 16;
            load_frag(pr, &scr[wave][0][pbase]);
            load_frag(pi, &scr[wave][1][pbase]);
        }
#pragma unroll
        for (int nt = 0; nt < 4; ++nt) {
            // V^T fragment: contiguous 32B (16 keys) of dv-row (nt*16+l15)
            FragBF vr, vi;
            size_t g = baseT + (size_t)(nt * 16 + l15) * CS + kt2 * 16 + half * 16;
            load_frag(vr, &Vtr[g]);
            load_frag(vi, &Vti[g]);
            oRR[nt] = wmma_bf16(pr, vr, oRR[nt]);
            oII[nt] = wmma_bf16(pi, vi, oII[nt]);
            oRI[nt] = wmma_bf16(pr, vi, oRI[nt]);
            oRI[nt] = wmma_bf16(pi, vr, oRI[nt]);
        }
    }

    // cross-wave reduction of partial O (ds_add_f32)
#pragma unroll
    for (int nt = 0; nt < 4; ++nt) {
#pragma unroll
        for (int v = 0; v < 8; ++v) {
            int r = v + 8 * half;
            int c = nt * 16 + l15;
            atomicAdd(&lOr[r * 64 + c], oRR[nt][v] - oII[nt][v]);
            atomicAdd(&lOi[r * 64 + c], oRI[nt][v]);
        }
    }
    __syncthreads();

    // store O: [B,S,H*DV]
    const int bb = bh >> 4, hh = bh & 15;
    for (int i = tid; i < 16 * 64; i += 256) {
        int r = i >> 6, c = i & 63;
        size_t o = ((size_t)bb * CS + qb * 16 + r) * (CH * CDV) + hh * 64 + c;
        Or_[o] = f2bf(lOr[i]);
        Oi_[o] = f2bf(lOi[i]);
    }
}

// ---------------------------------------------------------------------------
// Residual + complex layernorm (2x2 covariance whitening).
// One workgroup per row (B*S rows), 256 threads, D=1024 -> 4 elems/thread.
// out[2, B, S, D] f32.
// ---------------------------------------------------------------------------
__global__ __launch_bounds__(256)
void k_cv_layernorm(const float* __restrict__ Yr, const float* __restrict__ Yi,
                    const float* __restrict__ Rr, const float* __restrict__ Ri,
                    const float* __restrict__ grr, const float* __restrict__ gri,
                    const float* __restrict__ gii, const float* __restrict__ btr,
                    const float* __restrict__ bti, float* __restrict__ out) {
    __shared__ float red[256];
    const int tid = threadIdx.x;
    const size_t off = (size_t)blockIdx.x * CD;
    const size_t plane = (size_t)CB * CS * CD;

    float xr[4], xi[4];
    float sr = 0.f, si = 0.f;
#pragma unroll
    for (int t = 0; t < 4; ++t) {
        int j = tid + t * 256;
        xr[t] = Yr[off + j] + Rr[off + j];
        xi[t] = Yi[off + j] + Ri[off + j];
        sr += xr[t];
        si += xi[t];
    }

    auto blockSum = [&](float v) -> float {
        red[tid] = v;
        __syncthreads();
        for (int s = 128; s > 0; s >>= 1) {
            if (tid < s) red[tid] += red[tid + s];
            __syncthreads();
        }
        float r = red[0];
        __syncthreads();
        return r;
    };

    const float invD = 1.0f / (float)CD;
    float mr = blockSum(sr) * invD;
    float mi = blockSum(si) * invD;

    float vrr = 0.f, vii = 0.f, vri = 0.f;
#pragma unroll
    for (int t = 0; t < 4; ++t) {
        float r = xr[t] - mr, i = xi[t] - mi;
        vrr += r * r;
        vii += i * i;
        vri += r * i;
    }
    vrr = blockSum(vrr) * invD + CEPS;
    vii = blockSum(vii) * invD + CEPS;
    vri = blockSum(vri) * invD;

    float s  = sqrtf(vrr * vii - vri * vri);
    float tt = sqrtf(vrr + vii + 2.0f * s);
    float inv = 1.0f / (s * tt);
    float Wrr = (vii + s) * inv;
    float Wii = (vrr + s) * inv;
    float Wri = -vri * inv;

#pragma unroll
    for (int t = 0; t < 4; ++t) {
        int j = tid + t * 256;
        float r = xr[t] - mr, i = xi[t] - mi;
        float or_ = Wrr * r + Wri * i;
        float oi  = Wri * r + Wii * i;
        out[off + j]         = grr[j] * or_ + gri[j] * oi + btr[j];
        out[plane + off + j] = gri[j] * or_ + gii[j] * oi + bti[j];
    }
}

// ---------------------------------------------------------------------------
// Host-side launcher
// ---------------------------------------------------------------------------
extern "C" void kernel_launch(void* const* d_in, const int* in_sizes, int n_in,
                              void* d_out, int out_size, void* d_ws, size_t ws_size,
                              hipStream_t stream) {
    (void)in_sizes; (void)n_in; (void)out_size; (void)ws_size;

    const float* q_r  = (const float*)d_in[0];
    const float* q_i  = (const float*)d_in[1];
    const float* k_r  = (const float*)d_in[2];
    const float* k_i  = (const float*)d_in[3];
    const float* v_r  = (const float*)d_in[4];
    const float* v_i  = (const float*)d_in[5];
    const float* wq_r = (const float*)d_in[6];
    const float* wq_i = (const float*)d_in[7];
    const float* wk_r = (const float*)d_in[8];
    const float* wk_i = (const float*)d_in[9];
    const float* wv_r = (const float*)d_in[10];
    const float* wv_i = (const float*)d_in[11];
    const float* fc_r = (const float*)d_in[12];
    const float* fc_i = (const float*)d_in[13];
    const float* grr  = (const float*)d_in[14];
    const float* gri  = (const float*)d_in[15];
    const float* gii  = (const float*)d_in[16];
    const float* btr  = (const float*)d_in[17];
    const float* bti  = (const float*)d_in[18];

    const size_t EL  = (size_t)CB * CS * CD;  // 4M elements
    const size_t WEL = (size_t)CD * CD;       // 1M elements

    bf16_t* W = (bf16_t*)d_ws;
    bf16_t *qrb = W,          *qib = W + EL,     *krb = W + 2 * EL;
    bf16_t *kib = W + 3 * EL, *vrb = W + 4 * EL, *vib = W + 5 * EL;
    bf16_t* wb = W + 6 * EL;
    bf16_t *wqr = wb,           *wqi = wb + WEL,     *wkr = wb + 2 * WEL;
    bf16_t *wki = wb + 3 * WEL, *wvr = wb + 4 * WEL, *wvi = wb + 5 * WEL;
    bf16_t *fcr = wb + 6 * WEL, *fci = wb + 7 * WEL;
    bf16_t* pb = wb + 8 * WEL;
    bf16_t *Qpr = pb,          *Qpi = pb + EL,     *Kpr = pb + 2 * EL;
    bf16_t *Kpi = pb + 3 * EL, *Vpr = pb + 4 * EL, *Vpi = pb + 5 * EL;
    bf16_t *Obr = pb + 6 * EL, *Obi = pb + 7 * EL;
    float* fl = (float*)(pb + 8 * EL);
    float *Yr = fl, *Yi = fl + EL;

    // 1) f32 -> bf16 conversions
    const float* srcs[14] = {q_r, q_i, k_r, k_i, v_r, v_i,
                             wq_r, wq_i, wk_r, wk_i, wv_r, wv_i, fc_r, fc_i};
    bf16_t* dsts[14] = {qrb, qib, krb, kib, vrb, vib,
                        wqr, wqi, wkr, wki, wvr, wvi, fcr, fci};
    for (int i = 0; i < 14; ++i) {
        size_t n = (i < 6) ? EL : WEL;
        k_f32_to_bf16<<<dim3((unsigned)((n + 255) / 256)), dim3(256), 0, stream>>>(
            srcs[i], dsts[i], (int)n);
    }

    // 2) complex QKV projections (M=4096, N=1024, K=1024)
    dim3 gg(64, 16);
    k_cgemm_bf16<<<gg, 256, 0, stream>>>(qrb, qib, wqr, wqi, Qpr, Qpi,
                                         nullptr, nullptr, CB * CS, CH * CDK, CD, 1);
    k_cgemm_bf16<<<gg, 256, 0, stream>>>(krb, kib, wkr, wki, Kpr, Kpi,
                                         nullptr, nullptr, CB * CS, CH * CDK, CD, 1);
    k_cgemm_bf16<<<gg, 256, 0, stream>>>(vrb, vib, wvr, wvi, Vpr, Vpi,
                                         nullptr, nullptr, CB * CS, CH * CDV, CD, 2);

    // 3) fused attention (B*H*(S/16) = 4096 workgroups)
    k_cattention<<<dim3(CB * CH * (CS / 16)), dim3(256), 0, stream>>>(
        Qpr, Qpi, Kpr, Kpi, Vpr, Vpi, Obr, Obi);

    // 4) output projection fc (f32 out for residual + LN)
    k_cgemm_bf16<<<gg, 256, 0, stream>>>(Obr, Obi, fcr, fci, nullptr, nullptr,
                                         Yr, Yi, CB * CS, CD, CH * CDV, 0);

    // 5) residual + complex layernorm -> d_out [2,B,S,D] f32
    k_cv_layernorm<<<dim3(CB * CS), dim3(256), 0, stream>>>(
        Yr, Yi, q_r, q_i, grr, gri, gii, btr, bti, (float*)d_out);
}
